// UpsamplerAttentionSingle_41918880808997
// MI455X (gfx1250) — matrix operand order repaired and checked by reading, not compiled, not run on hardware
//
#include <hip/hip_runtime.h>
#include <hip/hip_bf16.h>

#define B_  8
#define CC_ 64
#define CF_ 128
#define H_  128
#define W_  128
#define STR 136   // padded LDS row stride (bf16 elements), multiple of 8, bank-spreading

typedef __attribute__((ext_vector_type(16))) __bf16 v16bf;
typedef __attribute__((ext_vector_type(8)))  __bf16 v8bf;
typedef __attribute__((ext_vector_type(8)))  float  v8f;

// A-matrix 16x32 bf16 fragment (ISA 7.12.2): lane holds row M=lane%16,
// elements 0..7 -> K = kbase + 8*hi + (0..7); elements 8..15 -> K = kbase + 16 + 8*hi + (0..7)
__device__ __forceinline__ v16bf ld_a(const __bf16* row, int kbase, int hi) {
    v8bf lo = *reinterpret_cast<const v8bf*>(row + kbase + 8 * hi);
    v8bf hh = *reinterpret_cast<const v8bf*>(row + kbase + 16 + 8 * hi);
    return __builtin_shufflevector(lo, hh, 0,1,2,3,4,5,6,7,8,9,10,11,12,13,14,15);
}

// B-matrix 32x16 bf16 fragment: lane holds column N=lane%16,
// elements 0..7 -> K = kbase + 8*hi + (0..7); elements 8..15 -> K = kbase + 16 + 8*hi + (0..7)
__device__ __forceinline__ v16bf ld_b(const __bf16* row, int kbase, int hi) {
    v8bf lo = *reinterpret_cast<const v8bf*>(row + kbase + 8 * hi);
    v8bf hh = *reinterpret_cast<const v8bf*>(row + kbase + 16 + 8 * hi);
    return __builtin_shufflevector(lo, hh, 0,1,2,3,4,5,6,7,8,9,10,11,12,13,14,15);
}

// ---------------- k0: repack conv_w (CC,CF,3,3) fp32 -> bf16 [kh*3+kw][cc*CF+cf] ----------------
__global__ __launch_bounds__(256)
void k0_cw(const float* __restrict__ conv_w, __bf16* __restrict__ cw) {
    int idx = blockIdx.x * 256 + threadIdx.x;
    if (idx >= 9 * CC_ * CF_) return;
    int kk = idx / (CC_ * CF_);
    int r  = idx % (CC_ * CF_);
    int cc = r >> 7, cf = r & 127;
    int kh = kk / 3, kw = kk % 3;
    cw[idx] = (__bf16)conv_w[((size_t)(cc * CF_ + cf) * 3 + kh) * 3 + kw];
}

// ---------------- k1: proj = W_ctx^T @ m ; self_logit = (W_feat@m)·m + b_feat·m ; bdot = b_ctx·m --
__global__ __launch_bounds__(256)
void k1_proj_logits(const float* __restrict__ mainstream,
                    const float* __restrict__ W_ctx, const float* __restrict__ b_ctx,
                    const float* __restrict__ W_feat, const float* __restrict__ b_feat,
                    float* __restrict__ proj, float* __restrict__ S, float* __restrict__ T) {
    __shared__ alignas(16) __bf16 ms[W_ * STR];     // [pos][cf]  (B operand, column-per-lane)
    __shared__ alignas(16) __bf16 actx[CC_ * STR];  // [cc][cf]   (A = W_ctx^T)
    __shared__ alignas(16) __bf16 wfe[CF_ * STR];   // [d][c]     (A = W_feat)
    int b = blockIdx.x >> 7, h = blockIdx.x & 127;
    int tid = threadIdx.x;

    for (int idx = tid; idx < CF_ * W_; idx += 256) {       // coalesced over w
        int cf = idx >> 7, w = idx & 127;
        ms[w * STR + cf] = (__bf16)mainstream[(((size_t)b * CF_ + cf) * H_ + h) * W_ + w];
    }
    for (int idx = tid; idx < CF_ * CC_; idx += 256) {      // W_ctx is (CF,CC) row-major
        int cf = idx >> 6, cc = idx & 63;
        actx[cc * STR + cf] = (__bf16)W_ctx[idx];
    }
    for (int idx = tid; idx < CF_ * CF_; idx += 256) {
        int d = idx >> 7, c = idx & 127;
        wfe[d * STR + c] = (__bf16)W_feat[idx];
    }
    __syncthreads();

    int wv = tid >> 5, lane = tid & 31, hi = lane >> 4, lr = lane & 15;
    int n = wv * 16 + lr;  // spatial column owned by this lane

    // proj: 64 x 128 output, wave owns tile column n/16
#pragma unroll
    for (int tm = 0; tm < 4; ++tm) {
        v8f acc = {};
#pragma unroll
        for (int kc = 0; kc < 4; ++kc) {
            v16bf a  = ld_a(&actx[(tm * 16 + lr) * STR], kc * 32, hi);
            v16bf bb = ld_b(&ms[n * STR], kc * 32, hi);
            acc = __builtin_amdgcn_wmma_f32_16x16x32_bf16(false, a, false, bb,
                                                          (short)0, acc, false, false);
        }
        float* pr = proj + (((size_t)(b * H_ + h)) * W_ + n) * CC_;
#pragma unroll
        for (int r = 0; r < 8; ++r) pr[tm * 16 + r + 8 * hi] = acc[r];
    }

    // q = W_feat @ m, folded into column-wise dot with m (self logit)
    float sacc = 0.f;
#pragma unroll
    for (int tm = 0; tm < 8; ++tm) {
        v8f acc = {};
#pragma unroll
        for (int kc = 0; kc < 4; ++kc) {
            v16bf a  = ld_a(&wfe[(tm * 16 + lr) * STR], kc * 32, hi);
            v16bf bb = ld_b(&ms[n * STR], kc * 32, hi);
            acc = __builtin_amdgcn_wmma_f32_16x16x32_bf16(false, a, false, bb,
                                                          (short)0, acc, false, false);
        }
#pragma unroll
        for (int r = 0; r < 8; ++r) {
            int d = tm * 16 + r + 8 * hi;  // C/D layout: M = r + 8*(lane/16)
            sacc += acc[r] * (float)ms[n * STR + d];
        }
    }
    sacc += __shfl_xor(sacc, 16, 32);  // combine the two M-halves per column

    if (hi == 0) {
        float bfd = 0.f, bcd = 0.f;
        for (int c = 0; c < CF_; ++c) {
            float mv = (float)ms[n * STR + c];
            bfd += b_feat[c] * mv;
            bcd += b_ctx[c] * mv;
        }
        size_t o = (size_t)(b * H_ + h) * W_ + n;
        S[o] = sacc + bfd;  // self_logit
        T[o] = bcd;         // b_ctx · m  (ctx-logit bias)
    }
}

// ---------------- k2: values = conv3x3(mainstream) + conv_b, implicit GEMM ----------------
__global__ __launch_bounds__(256)
void k2_conv(const float* __restrict__ mainstream, const __bf16* __restrict__ cw,
             const float* __restrict__ conv_b, float* __restrict__ val) {
    __shared__ alignas(16) __bf16 msr[(W_ + 2) * STR];  // padded row [p=0..129][cf]
    __shared__ alignas(16) __bf16 aw[3][CC_ * STR];     // 3 kw taps of current kh
    int b = blockIdx.x >> 7, h = blockIdx.x & 127;
    int tid = threadIdx.x;
    int wv = tid >> 5, lane = tid & 31, hi = lane >> 4, lr = lane & 15;
    int n = wv * 16 + lr;

    v8f acc[4] = {};
    for (int kh = 0; kh < 3; ++kh) {
        int hs = h + kh - 1;
        __syncthreads();
        for (int idx = tid; idx < (W_ + 2) * CF_; idx += 256) {
            int cf = idx / (W_ + 2), p = idx % (W_ + 2);   // coalesced over p
            int ws = p - 1;
            float f = 0.f;
            if (hs >= 0 && hs < H_ && ws >= 0 && ws < W_)
                f = mainstream[(((size_t)b * CF_ + cf) * H_ + hs) * W_ + ws];
            msr[p * STR + cf] = (__bf16)f;
        }
        for (int kw = 0; kw < 3; ++kw)
            for (int idx = tid; idx < CC_ * CF_; idx += 256) {
                int cc = idx >> 7, cf = idx & 127;
                aw[kw][cc * STR + cf] = cw[(size_t)(kh * 3 + kw) * CC_ * CF_ + idx];
            }
        __syncthreads();
#pragma unroll
        for (int kw = 0; kw < 3; ++kw) {
#pragma unroll
            for (int tm = 0; tm < 4; ++tm) {
#pragma unroll
                for (int kc = 0; kc < 4; ++kc) {
                    v16bf a  = ld_a(&aw[kw][(tm * 16 + lr) * STR], kc * 32, hi);
                    v16bf bb = ld_b(&msr[(n + kw) * STR], kc * 32, hi);
                    acc[tm] = __builtin_amdgcn_wmma_f32_16x16x32_bf16(false, a, false, bb,
                                                                      (short)0, acc[tm], false, false);
                }
            }
        }
    }
    float* vr = val + (((size_t)(b * H_ + h)) * W_ + n) * CC_;
#pragma unroll
    for (int tm = 0; tm < 4; ++tm)
#pragma unroll
        for (int r = 0; r < 8; ++r) {
            int cc = tm * 16 + r + 8 * hi;
            vr[cc] = acc[tm][r] + conv_b[cc];
        }
}

// ---------------- k3: gate + blend (memory bound) ----------------
__global__ __launch_bounds__(256)
void k3_fuse(const float* __restrict__ contexts, const float* __restrict__ proj,
             const float* __restrict__ val, const float* __restrict__ S,
             const float* __restrict__ T, float* __restrict__ out) {
    __shared__ float projs[W_ * 68];
    __shared__ float vals[W_ * 68];
    __shared__ float Srow[W_];
    __shared__ float Trow[W_];
    int b = blockIdx.x >> 8, hf = blockIdx.x & 255;
    int h = hf >> 1;
    int tid = threadIdx.x;
    size_t rowoff = ((size_t)(b * H_ + h)) * W_;

    for (int idx = tid; idx < W_ * CC_; idx += 256) {
        int wc = idx >> 6, cc = idx & 63;
        projs[wc * 68 + cc] = proj[rowoff * CC_ + idx];
        vals[wc * 68 + cc]  = val[rowoff * CC_ + idx];
    }
    for (int idx = tid; idx < W_; idx += 256) {
        Srow[idx] = S[rowoff + idx];
        Trow[idx] = T[rowoff + idx];
    }
    __syncthreads();

    int wf = tid;        // fine column 0..255
    int wc = wf >> 1;    // coarse column
    float cv[CC_];
    float cl = 0.f;
#pragma unroll
    for (int cc = 0; cc < CC_; ++cc) {      // coalesced context stream
        float c = contexts[(((size_t)b * CC_ + cc) * (2 * H_) + hf) * (2 * W_) + wf];
        cv[cc] = c;
        cl += c * projs[wc * 68 + cc];
    }
    // 2-way softmax([ctx, self]) -> w0 = sigmoid(ctx - self)
    float l  = cl + Trow[wc] - Srow[wc];
    float w0 = 1.f / (1.f + __expf(-l));
    float w1 = 1.f - w0;
#pragma unroll
    for (int cc = 0; cc < CC_; ++cc) {      // coalesced output stream
        out[(((size_t)b * CC_ + cc) * (2 * H_) + hf) * (2 * W_) + wf]
            = w0 * cv[cc] + w1 * vals[wc * 68 + cc];
    }
}

extern "C" void kernel_launch(void* const* d_in, const int* in_sizes, int n_in,
                              void* d_out, int out_size, void* d_ws, size_t ws_size,
                              hipStream_t stream) {
    const float* contexts   = (const float*)d_in[0];
    const float* mainstream = (const float*)d_in[1];
    const float* W_ctx      = (const float*)d_in[2];
    const float* b_ctx      = (const float*)d_in[3];
    const float* W_feat     = (const float*)d_in[4];
    const float* b_feat     = (const float*)d_in[5];
    const float* conv_w     = (const float*)d_in[6];
    const float* conv_b     = (const float*)d_in[7];
    float* out = (float*)d_out;

    char* ws = (char*)d_ws;
    float* proj = (float*)ws; ws += (size_t)B_ * H_ * W_ * CC_ * sizeof(float);
    float* val  = (float*)ws; ws += (size_t)B_ * H_ * W_ * CC_ * sizeof(float);
    float* S    = (float*)ws; ws += (size_t)B_ * H_ * W_ * sizeof(float);
    float* T    = (float*)ws; ws += (size_t)B_ * H_ * W_ * sizeof(float);
    __bf16* cw  = (__bf16*)ws;

    k0_cw<<<(9 * CC_ * CF_ + 255) / 256, 256, 0, stream>>>(conv_w, cw);
    k1_proj_logits<<<B_ * H_, 256, 0, stream>>>(mainstream, W_ctx, b_ctx, W_feat, b_feat,
                                                proj, S, T);
    k2_conv<<<B_ * H_, 256, 0, stream>>>(mainstream, cw, conv_b, val);
    k3_fuse<<<B_ * 2 * H_, 256, 0, stream>>>(contexts, proj, val, S, T, out);
}
